// _ligru_cell_jit_17781164606050
// MI455X (gfx1250) — matrix operand order repaired and checked by reading, not compiled, not run on hardware
//
#include <hip/hip_runtime.h>
#include <hip/hip_bf16.h>

// ---------------------------------------------------------------------------
// LiGRU cell forward for MI455X (gfx1250), wave32 + WMMA bf16.
//
//   B=128, T=500, H=512.  Per step: gates = wx[:,t,:] + h @ uT  ([128,1024]),
//   then sigmoid/relu/lerp elementwise, streaming 4 outputs of [B,T,H].
//
// Strategy:
//   * 64 workgroups x 256 threads (8 wave32s). WG w owns hidden cols
//     [8w, 8w+8) -> gate cols {8w..8w+7} (at) and {512+8w..} (zt).
//   * Each wave owns one 16-row M-tile of the batch; per step it runs
//     16x v_wmma_f32_16x16x32_bf16 over K=512.
//   * B operand (u slice, 16 cols x 512 K) is loop-invariant: held in
//     128 VGPRs for the whole time loop (zero u traffic per step).
//   * h double buffer lives in d_ws as *bf16* (2 x 128 KB, L2-resident):
//     A fragments load bf16 directly (2x b128 per fragment, no cvt on the
//     read path).  Exact fp32 hidden state is kept in registers (hreg) by
//     the owning lanes; outputs are written from fp32.
//   * Grid-wide step barrier: monotonic atomic counter in d_ws, reset by
//     an init kernel every launch (graph-capture safe, no host sync).
// ---------------------------------------------------------------------------

typedef __attribute__((ext_vector_type(16))) __bf16 v16bf;
typedef __attribute__((ext_vector_type(8)))  __bf16 v8bf;
typedef __attribute__((ext_vector_type(8)))  float  v8f;

#define LG_B   128
#define LG_T   500
#define LG_H   512
#define LG_2H  1024
#define LG_NWG 64      // 1024 gate cols / 16 per WG
#define LG_OUT ((size_t)LG_B * LG_T * LG_H)   // one output tensor, elements

__global__ void ligru_init(const float* __restrict__ ht,
                           __bf16* __restrict__ hbuf,
                           unsigned* __restrict__ bar) {
  int i = blockIdx.x * blockDim.x + threadIdx.x;
  if (i == 0) *bar = 0u;
  if (i < LG_B * LG_H) hbuf[i] = (__bf16)ht[i];   // h double-buffer slot 0 = ht
}

__global__ void __launch_bounds__(256, 1)
ligru_main(const float* __restrict__ wx,     // [B, T, 2H]
           const float* __restrict__ u,      // [2H, H]  (row gcol = uT col gcol)
           const float* __restrict__ ht,     // [B, H]   initial hidden state
           const float* __restrict__ drop,   // [B, H]
           float* __restrict__ out,          // 4 x [B, T, H] concatenated
           __bf16* __restrict__ hws,         // 2 x [B, H] bf16 double buffer
           unsigned* __restrict__ bar) {
  const int wave = threadIdx.x >> 5;     // 0..7 : M-tile (16 batch rows each)
  const int lane = threadIdx.x & 31;
  const int n    = lane & 15;            // column within 16-wide gate slice
  const int hi   = lane >> 4;            // 0 or 1 (lane half)
  const int j0   = blockIdx.x * 8;       // first hidden column of this WG
  const int jcol = j0 + (n & 7);         // hidden column this lane updates
  // gate column feeding this lane's B fragment / wx add / C column:
  const int gcol = (n < 8) ? (j0 + n) : (LG_H + j0 + (n - 8));

  __bf16* hb0 = hws;                     // buffer read on even steps
  __bf16* hb1 = hws + LG_B * LG_H;

  // ---- B fragments: u-slice, fp32 -> bf16, held in VGPRs for all 500 steps.
  // B(32x16) layout: lane holds col n; element e <-> K = kk*32 + hi*16 + e.
  v16bf bfrag[16];
  {
    const float* urow = u + (size_t)gcol * LG_H;   // uT column gcol, contiguous in K
#pragma unroll
    for (int kk = 0; kk < 16; ++kk) {
      const float* p = urow + kk * 32 + hi * 16;
      v16bf t;
#pragma unroll
      for (int e = 0; e < 16; ++e) t[e] = (__bf16)p[e];
      bfrag[kk] = t;
    }
  }

  // ---- per-lane loop-invariant state for the elementwise phase.
  // C/D layout: element v <-> row r = wave*16 + hi*8 + v, column n.
  float dm[8], hreg[8];
#pragma unroll
  for (int v = 0; v < 8; ++v) {
    int r = wave * 16 + hi * 8 + v;
    dm[v]   = drop[(size_t)r * LG_H + jcol];
    hreg[v] = ht[(size_t)r * LG_H + jcol];   // exact fp32 state, register-held
  }

  const int arow = wave * 16 + n;        // A-matrix row this lane loads

  for (int t = 0; t < LG_T; ++t) {
    const __bf16* hread  = (t & 1) ? hb1 : hb0;
    __bf16*       hwrite = (t & 1) ? hb0 : hb1;

    // ---- gates slice = h @ uT-slice : 16 WMMAs over K=512, f32 accumulate.
    v8f c = {};
#pragma unroll
    for (int kk = 0; kk < 16; ++kk) {
      // A(16x32) layout: element e <-> K = kk*32 + hi*16 + e (matches B).
      // Both 8-element runs are 16B-aligned -> one global_load_b128 each.
      const __bf16* hr = hread + (size_t)arow * LG_H + kk * 32 + hi * 16;
      v8bf lo = *(const v8bf*)(hr);
      v8bf hv = *(const v8bf*)(hr + 8);
      v16bf a;
#pragma unroll
      for (int e = 0; e < 8; ++e) { a[e] = lo[e]; a[8 + e] = hv[e]; }
      c = __builtin_amdgcn_wmma_f32_16x16x32_bf16(
              /*neg_a=*/false, a, /*neg_b=*/false, bfrag[kk],
              /*c_mod=*/(short)0, c, /*reuse_a=*/false, /*reuse_b=*/false);
    }

    // ---- add wx, activations, state update, streaming stores.
    const float* wxt = wx + (size_t)t * LG_2H + gcol;
#pragma unroll
    for (int v = 0; v < 8; ++v) {
      int r = wave * 16 + hi * 8 + v;
      float g = c[v] + wxt[(size_t)r * LG_T * LG_2H];
      float p = __shfl_xor(g, 8, 32);          // swap at <-> zt across lane^8
      float atv = (n < 8) ? g : p;             // pre-activation 'at'
      float ztp = (n < 8) ? p : g;             // pre-sigmoid 'zt'
      float zt  = 1.0f / (1.0f + __expf(-ztp));
      float hc  = fmaxf(atv, 0.0f) * dm[v];    // relu * drop_mask
      float hn  = hreg[v] * zt + (1.0f - zt) * hc;
      hreg[v] = hn;
      if (n < 8) {                             // one writer per (r, jcol)
        size_t o = (size_t)r * LG_T * LG_H + (size_t)t * LG_H + jcol;
        out[o]              = hn;              // hs
        out[o + LG_OUT]     = zt;              // zs
        out[o + 2 * LG_OUT] = atv;             // ats
        out[o + 3 * LG_OUT] = hc;              // hcs
        hwrite[(size_t)r * LG_H + jcol] = (__bf16)hn;  // next-step h (bf16)
      }
    }

    // ---- grid-wide step barrier (monotonic counter; double buffer makes
    //      one barrier per step race-free).
    __threadfence();
    __syncthreads();
    if (threadIdx.x == 0) {
      atomicAdd(bar, 1u);
      unsigned target = (unsigned)(t + 1) * (unsigned)gridDim.x;
      while (atomicAdd(bar, 0u) < target) {
        __builtin_amdgcn_s_sleep(1);
      }
    }
    __syncthreads();
  }
}

extern "C" void kernel_launch(void* const* d_in, const int* in_sizes, int n_in,
                              void* d_out, int out_size, void* d_ws, size_t ws_size,
                              hipStream_t stream) {
  const float* wx   = (const float*)d_in[0];   // [B, T, 2H]
  const float* u    = (const float*)d_in[1];   // [2H, H]
  const float* ht   = (const float*)d_in[2];   // [B, H]
  const float* drop = (const float*)d_in[3];   // [B, H]
  float* out = (float*)d_out;

  __bf16*   hbuf = (__bf16*)d_ws;                                  // 2 x B*H bf16
  unsigned* bar  = (unsigned*)((char*)d_ws +
                               (size_t)2 * LG_B * LG_H * sizeof(__bf16));

  ligru_init<<<(LG_B * LG_H + 255) / 256, 256, 0, stream>>>(ht, hbuf, bar);
  ligru_main<<<LG_NWG, 256, 0, stream>>>(wx, u, ht, drop, out, hbuf, bar);
}
